// ManualRNN_48524540510892
// MI455X (gfx1250) — matrix operand order repaired and checked by reading, not compiled, not run on hardware
//
#include <hip/hip_runtime.h>

#define T_STEPS 2048
#define IN_SZ   64
#define HID     128
#define OUT_SZ  64
#define BATCH   256

typedef __attribute__((ext_vector_type(16))) _Float16 v16h;
typedef __attribute__((ext_vector_type(8)))  float    v8f;

union AH { v16h v; unsigned int u[8]; };

__device__ inline unsigned short f2h(float f) {
    _Float16 h = (_Float16)f;
    return __builtin_bit_cast(unsigned short, h);
}
__device__ inline unsigned int pack2h(float a, float b) {
    return (unsigned int)f2h(a) | ((unsigned int)f2h(b) << 16);
}
__device__ inline float fast_tanh(float x) {
    x = fminf(fmaxf(x, -15.0f), 15.0f);
    float e = __expf(2.0f * x);
    return (e - 1.0f) / (e + 1.0f);
}

// One workgroup = 16 batch rows, 8 waves each owning 16 columns of H.
// Weights stay in registers as WMMA B-matrices; h round-trips through LDS
// (f16, double-buffered, one barrier/step); x prefetched 2 steps ahead.
__global__ __launch_bounds__(256) void rnn_scan_kernel(
    const float* __restrict__ x, const float* __restrict__ Wx_w,
    const float* __restrict__ Wx_b, const float* __restrict__ Wh_w,
    const float* __restrict__ Wh_b, float* __restrict__ h_out)
{
    __shared__ unsigned int hbuf[2][16 * 65]; // 128 f16/row + 1-dword pad
    __shared__ unsigned int xbuf[2][16 * 33]; //  64 f16/row + 1-dword pad

    const int tid  = threadIdx.x;
    const int lane = tid & 31;
    const int wave = tid >> 5;
    const int b0   = blockIdx.x * 16;
    const int n0   = wave * 16;
    const int n    = lane & 15;
    const int g    = lane >> 4;

    // ---- preload weights into B-matrix registers (f16) ----
    // B[k][ncol] = W[n0+n][k]; lane holds N=n, K-group by g, pairs per VGPR.
    v16h Bh[4], Bx[2];
#pragma unroll
    for (int c = 0; c < 4; ++c) {
        AH tb;
#pragma unroll
        for (int v = 0; v < 8; ++v) {
            int k = 32 * c + 16 * g + 2 * v;
            const float* p = Wh_w + (size_t)(n0 + n) * HID + k;
            tb.u[v] = pack2h(p[0], p[1]);
        }
        Bh[c] = tb.v;
    }
#pragma unroll
    for (int c = 0; c < 2; ++c) {
        AH tb;
#pragma unroll
        for (int v = 0; v < 8; ++v) {
            int k = 32 * c + 16 * g + 2 * v;
            const float* p = Wx_w + (size_t)(n0 + n) * IN_SZ + k;
            tb.u[v] = pack2h(p[0], p[1]);
        }
        Bx[c] = tb.v;
    }
    const float biasv = Wx_b[n0 + n] + Wh_b[n0 + n];

    // h_{-1} = 0 lives in parity-1 buffer
    for (int i = tid; i < 16 * 65; i += 256) hbuf[1][i] = 0u;

    // ---- stage x_0 into xbuf[0]; prefetch x_1 into registers ----
    const int xr = tid >> 4;         // batch row within tile 0..15
    const int xi = (tid & 15) * 4;   // 4 consecutive floats per thread
    const float* xrow = x + ((size_t)(b0 + xr) * T_STEPS) * IN_SZ + xi;

    float4 f0 = *(const float4*)(xrow);
    xbuf[0][xr * 33 + (xi >> 1)]     = pack2h(f0.x, f0.y);
    xbuf[0][xr * 33 + (xi >> 1) + 1] = pack2h(f0.z, f0.w);
    float4 xnext = *(const float4*)(xrow + (size_t)1 * IN_SZ);

    __syncthreads();

    float hval[8];
#pragma unroll 1
    for (int t = 0; t < T_STEPS; ++t) {
        const int cur = t & 1, prv = cur ^ 1;

        // stage prefetched x_{t+1}; issue load for x_{t+2}
        xbuf[prv][xr * 33 + (xi >> 1)]     = pack2h(xnext.x, xnext.y);
        xbuf[prv][xr * 33 + (xi >> 1) + 1] = pack2h(xnext.z, xnext.w);
        int tp = t + 2; if (tp > T_STEPS - 1) tp = T_STEPS - 1;
        xnext = *(const float4*)(xrow + (size_t)tp * IN_SZ);

        v8f acc;
#pragma unroll
        for (int i = 0; i < 8; ++i) acc[i] = biasv;

        // x-projection: K = 64 in 2 chunks of 32
#pragma unroll
        for (int c = 0; c < 2; ++c) {
            AH a;
#pragma unroll
            for (int v = 0; v < 8; ++v) {
                int dw = 16 * c + (g << 3) + v;
                a.u[v] = xbuf[cur][n * 33 + dw];
            }
            acc = __builtin_amdgcn_wmma_f32_16x16x32_f16(
                false, a.v, false, Bx[c], (short)0, acc, false, false);
        }
        // recurrence: K = 128 in 4 chunks of 32 over h_{t-1}
#pragma unroll
        for (int c = 0; c < 4; ++c) {
            AH a;
#pragma unroll
            for (int v = 0; v < 8; ++v) {
                int dw = 16 * c + (g << 3) + v;
                a.u[v] = hbuf[prv][n * 65 + dw];
            }
            acc = __builtin_amdgcn_wmma_f32_16x16x32_f16(
                false, a.v, false, Bh[c], (short)0, acc, false, false);
        }

        // tanh, store h_t (f16) for next step; D layout: M = v + 8g, N = n
        unsigned short* hb16 = (unsigned short*)&hbuf[cur][0];
#pragma unroll
        for (int v = 0; v < 8; ++v) {
            float hv = fast_tanh(acc[v]);
            hval[v] = hv;
            int m = v + (g << 3);
            hb16[m * 130 + (n0 + n)] = f2h(hv);
        }
        __syncthreads();
    }

    // final h (f32) -> workspace
#pragma unroll
    for (int v = 0; v < 8; ++v) {
        int m = v + (g << 3);
        h_out[(size_t)(b0 + m) * HID + n0 + n] = hval[v];
    }
}

// out = softmax(h_final @ fc_w^T + fc_b); one block per batch row.
__global__ __launch_bounds__(64) void fc_softmax_kernel(
    const float* __restrict__ h, const float* __restrict__ fc_w,
    const float* __restrict__ fc_b, float* __restrict__ out)
{
    __shared__ float hrow[HID];
    __shared__ float val[OUT_SZ];
    const int b = blockIdx.x;
    const int j = threadIdx.x;
    hrow[j]      = h[(size_t)b * HID + j];
    hrow[j + 64] = h[(size_t)b * HID + j + 64];
    __syncthreads();
    float s = fc_b[j];
#pragma unroll 4
    for (int k = 0; k < HID; ++k) s += hrow[k] * fc_w[(size_t)j * HID + k];
    val[j] = s;
    __syncthreads();
    float mx = -1e30f;
    for (int k = 0; k < OUT_SZ; ++k) mx = fmaxf(mx, val[k]);
    float e = __expf(s - mx);
    __syncthreads();
    val[j] = e;
    __syncthreads();
    float sum = 0.0f;
    for (int k = 0; k < OUT_SZ; ++k) sum += val[k];
    out[(size_t)b * OUT_SZ + j] = e / sum;
}

extern "C" void kernel_launch(void* const* d_in, const int* in_sizes, int n_in,
                              void* d_out, int out_size, void* d_ws, size_t ws_size,
                              hipStream_t stream) {
    const float* x    = (const float*)d_in[0];
    const float* Wx_w = (const float*)d_in[1];
    const float* Wx_b = (const float*)d_in[2];
    const float* Wh_w = (const float*)d_in[3];
    const float* Wh_b = (const float*)d_in[4];
    const float* fc_w = (const float*)d_in[5];
    const float* fc_b = (const float*)d_in[6];
    float* out = (float*)d_out;
    float* hws = (float*)d_ws; // 256*128 f32 = 128 KB

    rnn_scan_kernel<<<BATCH / 16, 256, 0, stream>>>(x, Wx_w, Wx_b, Wh_w, Wh_b, hws);
    fc_softmax_kernel<<<BATCH, 64, 0, stream>>>(hws, fc_w, fc_b, out);
}
